// SA_84310208020820
// MI455X (gfx1250) — matrix operand order repaired and checked by reading, not compiled, not run on hardware
//
#include <hip/hip_runtime.h>
#include <hip/hip_bf16.h>
#include <cstdint>

typedef _Float16 v16h __attribute__((ext_vector_type(16)));
typedef _Float16 v8h  __attribute__((ext_vector_type(8)));
typedef float    v8f  __attribute__((ext_vector_type(8)));
typedef unsigned int u32x4 __attribute__((ext_vector_type(4)));
typedef int i32x4 __attribute__((ext_vector_type(4)));
typedef int i32x8 __attribute__((ext_vector_type(8)));

#define NPOS  4096
#define BATCH 8
#define CDIM  64
#define DK    32
#define NTILE 64

// ---------------- TDM (Tensor Data Mover) helpers ----------------
__device__ __forceinline__ unsigned lds_off(const void* p) {
  return (unsigned)(uintptr_t)p;   // generic LDS pointer: low 32 bits = LDS byte offset
}

__device__ __forceinline__ void tdm_load_2d(unsigned lds_addr, const void* gptr,
                                            unsigned row_elems, unsigned nrows,
                                            unsigned long long row_stride)
{
  unsigned long long ga = (unsigned long long)(uintptr_t)gptr;
  u32x4 g0;
  g0[0] = (unsigned)__builtin_amdgcn_readfirstlane(1);                 // count=1
  g0[1] = (unsigned)__builtin_amdgcn_readfirstlane((int)lds_addr);
  g0[2] = (unsigned)__builtin_amdgcn_readfirstlane((int)(unsigned)ga);
  g0[3] = (unsigned)__builtin_amdgcn_readfirstlane(
            (int)(((unsigned)(ga >> 32) & 0x01FFFFFFu) | (2u << 30))); // addr[56:32] | type=2
  i32x8 g1;
  g1[0] = __builtin_amdgcn_readfirstlane((int)(1u << 16));             // data_size = 2B
  g1[1] = __builtin_amdgcn_readfirstlane((int)((row_elems & 0xFFFFu) << 16));
  g1[2] = __builtin_amdgcn_readfirstlane((int)(((row_elems >> 16) & 0xFFFFu)
                                               | ((nrows & 0xFFFFu) << 16)));
  g1[3] = __builtin_amdgcn_readfirstlane((int)(((nrows >> 16) & 0xFFFFu)
                                               | ((row_elems & 0xFFFFu) << 16)));  // tile_dim0
  g1[4] = __builtin_amdgcn_readfirstlane((int)(nrows & 0xFFFFu));      // tile_dim1
  g1[5] = __builtin_amdgcn_readfirstlane((int)(unsigned)(row_stride & 0xFFFFFFFFull));
  g1[6] = __builtin_amdgcn_readfirstlane((int)(unsigned)((row_stride >> 32) & 0xFFFFull));
  g1[7] = 0;
  i32x4 z4 = {0, 0, 0, 0};
#if __clang_major__ >= 23
  i32x8 z8 = {0, 0, 0, 0, 0, 0, 0, 0};
  __builtin_amdgcn_tensor_load_to_lds(g0, g1, z4, z4, z8, 0);
#else
  __builtin_amdgcn_tensor_load_to_lds(g0, g1, z4, z4, 0);
#endif
}

// ---------------- DPP 16-lane row reductions (no LDS, co-executes w/ WMMA) ---
template <int CTRL>
__device__ __forceinline__ float dpp_mov(float x) {
  int r = __builtin_amdgcn_update_dpp(0, __builtin_bit_cast(int, x),
                                      CTRL, 0xF, 0xF, true);
  return __builtin_bit_cast(float, r);
}
__device__ __forceinline__ float red_max16(float x) {
  x = fmaxf(x, dpp_mov<0xB1>(x));    // quad_perm [1,0,3,2]  (xor 1)
  x = fmaxf(x, dpp_mov<0x4E>(x));    // quad_perm [2,3,0,1]  (xor 2)
  x = fmaxf(x, dpp_mov<0x141>(x));   // row_half_mirror      (combine quads)
  x = fmaxf(x, dpp_mov<0x140>(x));   // row_mirror           (combine 8-groups)
  return x;
}
__device__ __forceinline__ float red_add16(float x) {
  x += dpp_mov<0xB1>(x);
  x += dpp_mov<0x4E>(x);
  x += dpp_mov<0x141>(x);
  x += dpp_mov<0x140>(x);
  return x;
}

// ---------------- matrix fragment loaders ----------------
// A-matrix 16x32 f16: lane = m%16; lanes<16: k 0..15, lanes>=16: k 16..31
__device__ __forceinline__ v16h load_amat(const _Float16* row, int hs) {
  v8h lo = *(const v8h*)(row + hs * 16);
  v8h hi = *(const v8h*)(row + hs * 16 + 8);
  v16h a;
#pragma unroll
  for (int i = 0; i < 8; ++i) { a[i] = lo[i]; a[8 + i] = hi[i]; }
  return a;
}
// B-matrix 32x16 f16: lane = column; 16 contiguous K values per lane
__device__ __forceinline__ v16h load_bmat(const _Float16* p) {
  v8h lo = *(const v8h*)(p), hi = *(const v8h*)(p + 8);
  v16h b;
#pragma unroll
  for (int i = 0; i < 8; ++i) { b[i] = lo[i]; b[8 + i] = hi[i]; }
  return b;
}

// ---------------- projection kernel: Q/K/V = conv1x1(x) -----------------
__global__ __launch_bounds__(256, 1) void SA_proj_kernel(
    const float* __restrict__ x,
    const float* __restrict__ Wb, const float* __restrict__ bb,
    const float* __restrict__ Wc, const float* __restrict__ bc,
    const float* __restrict__ Wd, const float* __restrict__ bd,
    _Float16* __restrict__ Qh, _Float16* __restrict__ Kh, _Float16* __restrict__ Vh)
{
  __shared__ __align__(16) float xs[64][65];
  const int b  = blockIdx.y;
  const int n0 = blockIdx.x * 64;

  for (int i = threadIdx.x; i < 64 * 64; i += 256) {
    int cin = i >> 6, j = i & 63;
    xs[cin][j] = x[((size_t)b * CDIM + cin) * NPOS + n0 + j];
  }
  __syncthreads();

  const int j = threadIdx.x & 63;
  const int quarter = threadIdx.x >> 6;   // uniform per wave

  if (quarter < 2) {
    const float* W    = quarter ? Wc : Wb;
    const float* bias = quarter ? bc : bb;
    _Float16* dst = (quarter ? Kh : Qh) + ((size_t)b * NPOS + n0 + j) * DK;
    _Float16 o[32];
#pragma unroll 4
    for (int r = 0; r < 32; ++r) {
      float acc = bias[r];
      const float* wr = W + r * 64;
#pragma unroll
      for (int c = 0; c < 64; ++c) acc = fmaf(wr[c], xs[c][j], acc);
      o[r] = (_Float16)acc;
    }
#pragma unroll
    for (int k = 0; k < 4; ++k) {
      v8h t;
#pragma unroll
      for (int i = 0; i < 8; ++i) t[i] = o[k * 8 + i];
      *(v8h*)(dst + k * 8) = t;
    }
  } else {
    const int rbase = (quarter - 2) * 32;
#pragma unroll 4
    for (int r2 = 0; r2 < 32; ++r2) {
      int r = rbase + r2;
      float acc = bd[r];
      const float* wr = Wd + r * 64;
#pragma unroll
      for (int c = 0; c < 64; ++c) acc = fmaf(wr[c], xs[c][j], acc);
      Vh[((size_t)b * CDIM + r) * NPOS + n0 + j] = (_Float16)acc;
    }
  }
}

// ---------------- flash-attention kernel ----------------
// grid = B*(N/64) blocks, 128 threads (4 waves); wave owns 16 m-rows,
// iterates over n in 64-column tiles (12 WMMAs / tile).
__global__ __launch_bounds__(128, 1) void SA_flash_kernel(
    const _Float16* __restrict__ Qh, const _Float16* __restrict__ Kh,
    const _Float16* __restrict__ Vh, const float* __restrict__ xin,
    float* __restrict__ out)
{
  __shared__ __align__(16) _Float16 ldsK[2][NTILE * 32];   // [n][c], 4 KB each
  __shared__ __align__(16) _Float16 ldsV[2][CDIM * NTILE]; // [c][n], 8 KB each
  __shared__ __align__(16) _Float16 ldsP[4][16 * NTILE];   // per-wave P, 2 KB each

  const int b     = blockIdx.x >> 6;
  const int mtile = blockIdx.x & 63;
  const int wave  = threadIdx.x >> 5;
  const int lane  = threadIdx.x & 31;
  const int hs    = lane >> 4;
  const int l15   = lane & 15;
  const int m_base = mtile * 64 + wave * 16;

  // Q tile as WMMA A-matrix (16x32 f16)
  const _Float16* qrow = Qh + ((size_t)b * NPOS + (m_base + l15)) * DK;
  v16h Aq = load_amat(qrow, hs);

  float row_max[8], row_sum[8];
#pragma unroll
  for (int v = 0; v < 8; ++v) { row_max[v] = -3.0e38f; row_sum[v] = 0.f; }
  v8f acc[4];
#pragma unroll
  for (int t = 0; t < 4; ++t) acc[t] = (v8f){};

  const _Float16* Kbase = Kh + (size_t)b * NPOS * DK;
  const _Float16* Vbase = Vh + (size_t)b * CDIM * NPOS;

  // prologue: DMA tile 0 into buffer 0 (single wave issues; TDM ignores EXEC)
  if (wave == 0) {
    tdm_load_2d(lds_off(&ldsK[0][0]), Kbase, NTILE * 32u, 1u, NTILE * 32u);
    tdm_load_2d(lds_off(&ldsV[0][0]), Vbase, NTILE, CDIM, (unsigned long long)NPOS);
  }

  const int NT = NPOS / NTILE;   // 64 iterations
  for (int it = 0; it < NT; ++it) {
    const int buf = it & 1;
    if (wave == 0) __builtin_amdgcn_s_wait_tensorcnt((short)0);
    __syncthreads();
    if (it + 1 < NT && wave == 0) {
      const int nb = (it + 1) * NTILE;
      tdm_load_2d(lds_off(&ldsK[buf ^ 1][0]), Kbase + (size_t)nb * DK,
                  NTILE * 32u, 1u, NTILE * 32u);
      tdm_load_2d(lds_off(&ldsV[buf ^ 1][0]), Vbase + nb,
                  NTILE, CDIM, (unsigned long long)NPOS);
    }

    // ---- S = Q K^T : four 16x16 WMMAs over 64 columns ----
    v8f s[4];
#pragma unroll
    for (int jt = 0; jt < 4; ++jt) {
      v16h Bk = load_bmat(&ldsK[buf][(16 * jt + l15) * 32 + hs * 16]);
      v8f z = {};
      s[jt] = __builtin_amdgcn_wmma_f32_16x16x32_f16(false, Aq, false, Bk,
                                                     (short)0, z, false, false);
    }

    // ---- online softmax over the 64 new columns (DPP row reductions) ----
    float scl[8];
#pragma unroll
    for (int v = 0; v < 8; ++v) {
      float mv = fmaxf(fmaxf(s[0][v], s[1][v]), fmaxf(s[2][v], s[3][v]));
      mv = red_max16(mv);
      float nm = fmaxf(row_max[v], mv);
      scl[v] = __expf(row_max[v] - nm);
      row_max[v] = nm;
      float ps = 0.f;
#pragma unroll
      for (int jt = 0; jt < 4; ++jt) {
        float e = __expf(s[jt][v] - nm);
        s[jt][v] = e;
        ps += e;
      }
      ps = red_add16(ps);
      row_sum[v] = row_sum[v] * scl[v] + ps;
    }
#pragma unroll
    for (int t = 0; t < 4; ++t)
#pragma unroll
      for (int v = 0; v < 8; ++v) acc[t][v] *= scl[v];

    // ---- P: D-layout -> A-layout via per-wave LDS patch ----
    {
      _Float16* pw = &ldsP[wave][0];
#pragma unroll
      for (int v = 0; v < 8; ++v) {
        int r = v + hs * 8;
#pragma unroll
        for (int jt = 0; jt < 4; ++jt)
          pw[r * NTILE + 16 * jt + l15] = (_Float16)s[jt][v];
      }
    }
    const _Float16* pr = &ldsP[wave][l15 * NTILE];   // same-wave DS is in-order
    v16h Ap0 = load_amat(pr, hs);        // k = n_local 0..31
    v16h Ap1 = load_amat(pr + 32, hs);   // k = n_local 32..63

    // ---- O += P V : four c-tiles x two k-halves ----
#pragma unroll
    for (int t = 0; t < 4; ++t) {
      const _Float16* vrow = &ldsV[buf][(16 * t + l15) * NTILE + hs * 16];
      v16h Bv0 = load_bmat(vrow);
      acc[t] = __builtin_amdgcn_wmma_f32_16x16x32_f16(false, Ap0, false, Bv0,
                                                      (short)0, acc[t], false, false);
      v16h Bv1 = load_bmat(vrow + 32);
      acc[t] = __builtin_amdgcn_wmma_f32_16x16x32_f16(false, Ap1, false, Bv1,
                                                      (short)0, acc[t], false, false);
    }
    __syncthreads();   // protect buf before TDM overwrites it two tiles later
  }

  // ---- normalize, add residual, store ----
  float inv[8];
#pragma unroll
  for (int v = 0; v < 8; ++v) inv[v] = 1.0f / row_sum[v];
#pragma unroll
  for (int t = 0; t < 4; ++t) {
#pragma unroll
    for (int v = 0; v < 8; ++v) {
      int m = m_base + v + hs * 8;
      int c = 16 * t + l15;
      size_t idx = ((size_t)b * CDIM + c) * NPOS + m;
      out[idx] = acc[t][v] * inv[v] + xin[idx];
    }
  }
}

// ---------------- launch ----------------
extern "C" void kernel_launch(void* const* d_in, const int* in_sizes, int n_in,
                              void* d_out, int out_size, void* d_ws, size_t ws_size,
                              hipStream_t stream) {
  const float* x  = (const float*)d_in[0];
  const float* Wb = (const float*)d_in[1];
  const float* bb = (const float*)d_in[2];
  const float* Wc = (const float*)d_in[3];
  const float* bc = (const float*)d_in[4];
  const float* Wd = (const float*)d_in[5];
  const float* bd = (const float*)d_in[6];

  _Float16* Qh = (_Float16*)d_ws;                       // [B][N][32]
  _Float16* Kh = Qh + (size_t)BATCH * NPOS * DK;        // [B][N][32]
  _Float16* Vh = Kh + (size_t)BATCH * NPOS * DK;        // [B][64][N]

  SA_proj_kernel<<<dim3(NPOS / 64, BATCH), 256, 0, stream>>>(
      x, Wb, bb, Wc, bc, Wd, bd, Qh, Kh, Vh);
  SA_flash_kernel<<<dim3(BATCH * 64), 128, 0, stream>>>(
      Qh, Kh, Vh, x, (float*)d_out);
}
